// FastAttention_78709570667419
// MI455X (gfx1250) — matrix operand order repaired and checked by reading, not compiled, not run on hardware
//
#include <hip/hip_runtime.h>

#define S_LEN   4096
#define HIDDEN  2048
#define NHEADS  16
#define HDIM    128
#define BQ      128     // query rows per workgroup (8 waves x 16)
#define BK      32      // keys per block
#define NWAVES  8

typedef __attribute__((ext_vector_type(16))) __bf16 v16bf;
typedef __attribute__((ext_vector_type(8)))  float  v8f;

union BF16x16 { v16bf v; unsigned int u[8]; };

static __device__ __forceinline__ unsigned int pack_bf2(float lo, float hi) {
    unsigned short a = __builtin_bit_cast(unsigned short, (__bf16)lo);
    unsigned short b = __builtin_bit_cast(unsigned short, (__bf16)hi);
    return (unsigned int)a | ((unsigned int)b << 16);
}

static __device__ __forceinline__ float fast_exp2(float x) {
#if __has_builtin(__builtin_amdgcn_exp2f)
    return __builtin_amdgcn_exp2f(x);   // v_exp_f32
#else
    return exp2f(x);
#endif
}

// ds_swizzle group-of-32 lane-XOR: offset = (xor<<10) | and(0x1f)
template <int IMM>
static __device__ __forceinline__ float swz(float x) {
    return __builtin_bit_cast(float,
        __builtin_amdgcn_ds_swizzle(__builtin_bit_cast(int, x), IMM));
}
static __device__ __forceinline__ float rmax16(float x) {
    x = fmaxf(x, swz<0x041f>(x));   // xor 1
    x = fmaxf(x, swz<0x081f>(x));   // xor 2
    x = fmaxf(x, swz<0x101f>(x));   // xor 4
    x = fmaxf(x, swz<0x201f>(x));   // xor 8
    return x;
}
static __device__ __forceinline__ float rsum16(float x) {
    x += swz<0x041f>(x);
    x += swz<0x081f>(x);
    x += swz<0x101f>(x);
    x += swz<0x201f>(x);
    return x;
}

__global__ __launch_bounds__(256)
void fa_causal_bf16_wmma(const float* __restrict__ Q, const float* __restrict__ K,
                         const float* __restrict__ V, float* __restrict__ O)
{
    // K packed bf16 pairs [key][dpair], V transposed bf16 [d][key],
    // per-wave P f32 bounce buffer. Total 32 KB.
    __shared__ unsigned int sK[BK * (HDIM / 2)];   // 8 KB
    __shared__ __bf16       sV[HDIM * BK];         // 8 KB
    __shared__ float        sP[NWAVES][16 * BK];   // 16 KB

    const int qblk = blockIdx.x;      // 0..31
    const int h    = blockIdx.y;
    const int b    = blockIdx.z;
    const int tid  = threadIdx.x;
    const int wave = tid >> 5;
    const int lane = tid & 31;
    const int ln   = lane & 15;       // lane within half
    const int hi   = lane >> 4;       // half-wave select

    const int qb = qblk * BQ;
    const int q0 = qb + wave * 16;    // this wave's first query row

    const size_t base = (size_t)b * S_LEN * HIDDEN + (size_t)h * HDIM;
    const float* Qp = Q + base;
    const float* Kp = K + base;
    const float* Vp = V + base;
    float*       Op = O + base;

    const float kscale = 1.4426950408889634f * 0.08838834764831845f; // log2(e)/sqrt(128)

    // ---- Q tile resident in A-matrix layout, pre-scaled by log2(e)/sqrt(D) ----
    BF16x16 qa[4];
    {
        const float* qrow = Qp + (size_t)(q0 + ln) * HIDDEN;
        #pragma unroll
        for (int c = 0; c < 4; ++c) {
            #pragma unroll
            for (int v = 0; v < 8; ++v) {
                int d0 = 32 * c + 16 * hi + 2 * v;
                float2 f = *(const float2*)(qrow + d0);
                qa[c].u[v] = pack_bf2(f.x * kscale, f.y * kscale);
            }
        }
    }

    v8f acc[8];
    #pragma unroll
    for (int nt = 0; nt < 8; ++nt) acc[nt] = (v8f){};
    float mrow[8], lrow[8];
    #pragma unroll
    for (int r = 0; r < 8; ++r) { mrow[r] = -__builtin_inff(); lrow[r] = 0.0f; }

    const int nblocks = (qb + BQ) / BK;     // causal truncation for this WG
    for (int jb = 0; jb < nblocks; ++jb) {
        const int kb = jb * BK;

        __syncthreads();   // previous iteration's consumers done before restage
        // ---- stage K block: 32x128 f32 -> packed bf16 pairs [key][dpair] ----
        #pragma unroll
        for (int i = 0; i < 8; ++i) {
            int p   = tid + i * 256;
            int key = p >> 6;
            int dp  = p & 63;
            float2 f = *(const float2*)(Kp + (size_t)(kb + key) * HIDDEN + 2 * dp);
            sK[key * 64 + dp] = pack_bf2(f.x, f.y);
        }
        // ---- stage V block transposed: [d][key] bf16 ----
        #pragma unroll
        for (int i = 0; i < 8; ++i) {
            int p   = tid + i * 256;
            int key = p >> 6;
            int d   = 2 * (p & 63);
            float2 f = *(const float2*)(Vp + (size_t)(kb + key) * HIDDEN + d);
            sV[d * BK + key]       = (__bf16)f.x;
            sV[(d + 1) * BK + key] = (__bf16)f.y;
        }
        if (jb + 1 < nblocks) {  // hint next block into cache (global_prefetch_b8)
            const size_t nrow = (size_t)(kb + BK + (tid >> 3)) * HIDDEN + (tid & 7) * 16;
            __builtin_prefetch(Kp + nrow, 0, 1);
            __builtin_prefetch(Vp + nrow, 0, 1);
        }
        __syncthreads();

        if (kb <= q0 + 15) {   // wave-uniform: EXEC all-ones inside
            // ---- QK^T: two 16x16 score tiles, 4 independent accumulator chains ----
            BF16x16 kb0[4], kb1[4];
            #pragma unroll
            for (int c = 0; c < 4; ++c) {
                #pragma unroll
                for (int v = 0; v < 8; ++v) {
                    int dp = 16 * c + 8 * hi + v;          // d = 32c + 16hi + 2v
                    kb0[c].u[v] = sK[ln * 64 + dp];
                    kb1[c].u[v] = sK[(16 + ln) * 64 + dp];
                }
            }
            v8f s0a = (v8f){}, s0b = (v8f){}, s1a = (v8f){}, s1b = (v8f){};
            s0a = __builtin_amdgcn_wmma_f32_16x16x32_bf16(false, qa[0].v, false, kb0[0].v, (short)0, s0a, false, false);
            s1a = __builtin_amdgcn_wmma_f32_16x16x32_bf16(false, qa[0].v, false, kb1[0].v, (short)0, s1a, false, false);
            s0b = __builtin_amdgcn_wmma_f32_16x16x32_bf16(false, qa[1].v, false, kb0[1].v, (short)0, s0b, false, false);
            s1b = __builtin_amdgcn_wmma_f32_16x16x32_bf16(false, qa[1].v, false, kb1[1].v, (short)0, s1b, false, false);
            s0a = __builtin_amdgcn_wmma_f32_16x16x32_bf16(false, qa[2].v, false, kb0[2].v, (short)0, s0a, false, false);
            s1a = __builtin_amdgcn_wmma_f32_16x16x32_bf16(false, qa[2].v, false, kb1[2].v, (short)0, s1a, false, false);
            s0b = __builtin_amdgcn_wmma_f32_16x16x32_bf16(false, qa[3].v, false, kb0[3].v, (short)0, s0b, false, false);
            s1b = __builtin_amdgcn_wmma_f32_16x16x32_bf16(false, qa[3].v, false, kb1[3].v, (short)0, s1b, false, false);
            v8f s0 = s0a + s0b;
            v8f s1 = s1a + s1b;

            // ---- online softmax (scores already in base-2 domain) ----
            const bool needmask = (kb + BK - 1 > q0);
            float p0[8], p1[8];
            #pragma unroll
            for (int r = 0; r < 8; ++r) {
                float a  = s0[r];
                float c1 = s1[r];
                if (needmask) {
                    int qrow = q0 + 8 * hi + r;
                    if (kb + ln > qrow)      a  = -__builtin_inff();
                    if (kb + 16 + ln > qrow) c1 = -__builtin_inff();
                }
                float cm    = rmax16(fmaxf(a, c1));
                float mnew  = fmaxf(mrow[r], cm);
                float alpha = fast_exp2(mrow[r] - mnew);
                float e0    = fast_exp2(a - mnew);
                float e1    = fast_exp2(c1 - mnew);
                float rs    = rsum16(e0 + e1);
                lrow[r] = lrow[r] * alpha + rs;
                mrow[r] = mnew;
                p0[r] = e0; p1[r] = e1;
                #pragma unroll
                for (int nt = 0; nt < 8; ++nt) acc[nt][r] *= alpha;
            }

            // ---- P: C-layout -> A-layout via wave-private LDS bounce ----
            float* Pw = &sP[wave][0];
            #pragma unroll
            for (int r = 0; r < 8; ++r) {
                int m = 8 * hi + r;
                Pw[m * BK + ln]      = p0[r];
                Pw[m * BK + 16 + ln] = p1[r];
            }
#if __has_builtin(__builtin_amdgcn_s_wait_dscnt)
            __builtin_amdgcn_s_wait_dscnt(0);
#else
            asm volatile("s_wait_dscnt 0" ::: "memory");
#endif
            BF16x16 pa;
            {
                const float* prow = Pw + ln * BK;
                #pragma unroll
                for (int v = 0; v < 8; ++v) {
                    int k0 = 16 * hi + 2 * v;
                    float2 f = *(const float2*)(prow + k0);
                    pa.u[v] = pack_bf2(f.x, f.y);
                }
            }

            // ---- P x V: 8 independent WMMAs, K=32 contraction each ----
            #pragma unroll
            for (int nt = 0; nt < 8; ++nt) {
                BF16x16 vb;
                #pragma unroll
                for (int v = 0; v < 8; ++v) {
                    // column dim = 16*nt + ln ; key pair = 16*hi + 2v, +1
                    vb.u[v] = *(const unsigned int*)&sV[(16 * nt + ln) * BK + 16 * hi + 2 * v];
                }
                acc[nt] = __builtin_amdgcn_wmma_f32_16x16x32_bf16(false, pa.v, false, vb.v,
                                                                  (short)0, acc[nt], false, false);
            }
        }
    }

    // ---- normalize and store (C layout: lane=dim col, VGPR=row) ----
    #pragma unroll
    for (int r = 0; r < 8; ++r) {
        int   qrow = q0 + 8 * hi + r;
        float inv  = 1.0f / lrow[r];
        float* orow = Op + (size_t)qrow * HIDDEN;
        #pragma unroll
        for (int nt = 0; nt < 8; ++nt)
            orow[16 * nt + ln] = acc[nt][r] * inv;
    }
}

extern "C" void kernel_launch(void* const* d_in, const int* in_sizes, int n_in,
                              void* d_out, int out_size, void* d_ws, size_t ws_size,
                              hipStream_t stream) {
    const float* Q = (const float*)d_in[0];
    const float* K = (const float*)d_in[1];
    const float* V = (const float*)d_in[2];
    float*       O = (float*)d_out;
    const int B = in_sizes[0] / (S_LEN * HIDDEN);   // 2
    dim3 grid(S_LEN / BQ, NHEADS, B);
    dim3 block(256);
    hipLaunchKernelGGL(fa_causal_bf16_wmma, grid, block, 0, stream, Q, K, V, O);
}
